// SymbolDemodulationNetworkCE_74156905333314
// MI455X (gfx1250) — compile-verified, matrix operand order checked by reading, not run on hardware
//
#include <hip/hip_runtime.h>
#include <cstdint>
#include <cstddef>

// SymbolDemodulationNetworkCE — MI455X (gfx1250) implementation.
// 5 GEMMs via v_wmma_f32_16x16x32_bf16 (wave32), fused BN(batch stats)+ReLU between.
// Weights pre-swizzled to a WMMA-native layout; both GEMM tiles staged with
// CDNA5 async copies (global_load_async_to_lds_b128, ASYNCcnt) into
// double-buffered LDS: no ds_stores and no pack VALU in the hot loop.
//
// d_in order (setup_inputs dict order):
//  [0]=x(1024x4456 f32) [1]=W1 [2]=b1 [3]=W2 [4]=b2 [5]=W3 [6]=b3 [7]=W4 [8]=b4
//  [9]=W5(3840x7680) [10]=b5 [11]=g1 [12]=be1 [13]=g2 [14]=be2 [15]=g3 [16]=be3 [17]=g4 [18]=be4
// d_out = 1024*7680 f32.

#define BATCH 1024
#define DIN   4456
#define DINP  4480   // padded to multiple of 32 for BK
#define DH    3840
#define NOUT  7680
#define BN_EPS 1e-5f

typedef __attribute__((ext_vector_type(16))) __bf16 v16bf;
typedef __attribute__((ext_vector_type(8)))  float  v8f;

union FragU { unsigned u[8]; v16bf v; };

__device__ __forceinline__ unsigned short f2bf(float f) {
  unsigned u = __float_as_uint(f);
  unsigned r = u + 0x7fffu + ((u >> 16) & 1u);   // round-to-nearest-even
  return (unsigned short)(r >> 16);
}

// CDNA5 async copy: 16 bytes global -> LDS, per lane; tracked by ASYNCcnt.
__device__ __forceinline__ void async_cp16(unsigned lds_off, const void* g) {
  unsigned long long ga = (unsigned long long)(size_t)g;
  asm volatile("global_load_async_to_lds_b128 %0, %1, off"
               :: "v"(lds_off), "v"(ga)
               : "memory");
}
__device__ __forceinline__ void wait_async0() {
  asm volatile("s_wait_asynccnt 0x0" ::: "memory");
}

// ---------------------------------------------------------------------------
// fp32 -> bf16 with per-row column padding (x: 4456 -> 4480 cols)
__global__ __launch_bounds__(256)
void sdn_cvt_pad_cols(const float* __restrict__ src, unsigned short* __restrict__ dst,
                      int rows, int srcCols, int dstCols) {
  long i = (long)blockIdx.x * blockDim.x + threadIdx.x;
  if (i >= (long)rows * dstCols) return;
  int r = (int)(i / dstCols), c = (int)(i % dstCols);
  dst[i] = (c < srcCols) ? f2bf(src[(long)r * srcCols + c]) : (unsigned short)0;
}

// fp32 W[K][N] -> bf16 swizzled Wt[k/32][n][k%32] (zero-pad rows k>=Ktrue).
// 32x32 tile transpose through LDS: reads and writes both coalesced.
// grid = (N/32, Kpad/32), block = 256.
__global__ __launch_bounds__(256)
void sdn_cvt_swz(const float* __restrict__ src, unsigned short* __restrict__ dst,
                 int Ktrue, int N) {
  __shared__ unsigned short t[32 * 33];
  const int n0 = blockIdx.x * 32;
  const int k0 = blockIdx.y * 32;
#pragma unroll
  for (int e = 0; e < 4; ++e) {
    int idx = threadIdx.x + e * 256;
    int kk = idx >> 5, nn = idx & 31;
    int k = k0 + kk;
    float v = (k < Ktrue) ? src[(size_t)k * N + n0 + nn] : 0.0f;
    t[kk * 33 + nn] = f2bf(v);
  }
  __syncthreads();
  unsigned short* out = dst + (size_t)blockIdx.y * N * 32 + (size_t)n0 * 32;
#pragma unroll
  for (int e = 0; e < 4; ++e) {
    int idx = threadIdx.x + e * 256;   // idx = nn*32 + kk (contiguous out)
    int nn = idx >> 5, kk = idx & 31;
    out[idx] = t[kk * 33 + nn];
  }
}

// ---------------------------------------------------------------------------
// Fragment gather from LDS, per CDNA5 ISA 16-bit A/B layout (16x32 / 32x16):
// lane 0-15: row = lane&15, kb = 0 ; lane 16-31: kb = 8.
// halves 0..7 -> K = kb+h, halves 8..15 -> K = kb+16+(h-8); pairs are consecutive K
// so each pair is one aligned b32 LDS read (compiler fuses to ds_load_2addr_b32).
// Row stride = 40 halves (80 B): keeps async b128 chunks 16B-aligned AND the
// 16-lane gather conflict-free (20*r mod 64 distinct for r=0..15).
#define LDS_STRIDE 40

__device__ __forceinline__ v16bf ld_frag(const unsigned short* tile, int row, int kb) {
  FragU f;
  const unsigned short* p = tile + row * LDS_STRIDE + kb;
#pragma unroll
  for (int j = 0; j < 4; ++j) f.u[j]     = *(const unsigned*)(p + 2 * j);
#pragma unroll
  for (int j = 0; j < 4; ++j) f.u[4 + j] = *(const unsigned*)(p + 16 + 2 * j);
  return f.v;
}

// ---------------------------------------------------------------------------
// GEMM: C[M,N] = A[M,K](bf16 row-major) * W(bf16 swizzled [K/32][N][32]) + bias[N]
// Block = 256 threads = 8 waves; tile 128x128x32; wave tile 32x64 (2x4 wmma);
// double-buffered LDS, async global->LDS staging, one barrier per K-step.
__global__ __launch_bounds__(256)
void sdn_gemm_bf16_wmma(const unsigned short* __restrict__ A,
                        const unsigned short* __restrict__ Bt,
                        const float* __restrict__ bias,
                        float* __restrict__ C,
                        int M, int N, int K) {
  __shared__ __align__(16) unsigned short Asm[2][128 * LDS_STRIDE];
  __shared__ __align__(16) unsigned short Bsm[2][128 * LDS_STRIDE];

  const int tid   = threadIdx.x;
  const int lane  = tid & 31;
  const int wave  = tid >> 5;
  const int m0    = blockIdx.y * 128;
  const int nb    = blockIdx.x * 128;
  const int waveM = (wave & 3) * 32;   // 4 waves down M
  const int waveN = (wave >> 2) * 64;  // 2 waves across N

  v8f acc[2][4];
#pragma unroll
  for (int mi = 0; mi < 2; ++mi)
#pragma unroll
    for (int ni = 0; ni < 4; ++ni)
#pragma unroll
      for (int i = 0; i < 8; ++i) acc[mi][ni][i] = 0.0f;

  // staging map: thread owns two 16B chunks of each 128x32 tile
  const int row  = tid >> 1;            // 0..127 (m for A, n for B)
  const int cofh = (tid & 1) * 16;      // halves offset within row: 0 or 16

  const unsigned short* gA = A + (size_t)(m0 + row) * K + cofh;        // + k0
  const unsigned short* gB = Bt + (size_t)(nb + row) * 32 + cofh;      // + (k0/32)*N*32
  const size_t stepB = (size_t)N * 32;

  unsigned ldsA[2], ldsB[2];
  ldsA[0] = (unsigned)(size_t)(const void*)(Asm[0] + row * LDS_STRIDE + cofh);
  ldsA[1] = (unsigned)(size_t)(const void*)(Asm[1] + row * LDS_STRIDE + cofh);
  ldsB[0] = (unsigned)(size_t)(const void*)(Bsm[0] + row * LDS_STRIDE + cofh);
  ldsB[1] = (unsigned)(size_t)(const void*)(Bsm[1] + row * LDS_STRIDE + cofh);

  auto stage_async = [&](int buf, int k0) {
    const unsigned short* a = gA + k0;
    const unsigned short* b = gB + (size_t)(k0 >> 5) * stepB;
    async_cp16(ldsA[buf],      a);
    async_cp16(ldsA[buf] + 16, a + 8);
    async_cp16(ldsB[buf],      b);
    async_cp16(ldsB[buf] + 16, b + 8);
  };

  const int rlane = lane & 15;
  const int kb    = (lane < 16) ? 0 : 8;

  stage_async(0, 0);
  wait_async0();
  __syncthreads();

  int cur = 0;
  for (int k0 = 0; k0 < K; k0 += 32) {
    const bool more = (k0 + 32 < K);
    if (more) stage_async(cur ^ 1, k0 + 32);  // async engine fills other buffer

    v16bf afr[2], bfr[4];
#pragma unroll
    for (int mi = 0; mi < 2; ++mi)
      afr[mi] = ld_frag(Asm[cur], waveM + mi * 16 + rlane, kb);
#pragma unroll
    for (int ni = 0; ni < 4; ++ni)
      bfr[ni] = ld_frag(Bsm[cur], waveN + ni * 16 + rlane, kb);

#pragma unroll
    for (int mi = 0; mi < 2; ++mi)
#pragma unroll
      for (int ni = 0; ni < 4; ++ni)
        acc[mi][ni] = __builtin_amdgcn_wmma_f32_16x16x32_bf16(
            false, afr[mi], false, bfr[ni], (short)0, acc[mi][ni], false, false);

    if (more) {
      wait_async0();    // own async writes done; barrier makes all visible
      __syncthreads();
      cur ^= 1;
    }
  }

  // epilogue: C/D layout -> lane gives N, vgpr i gives M=i (lanes0-15) / i+8 (16-31)
  const int mhi = (lane >> 4) * 8;
#pragma unroll
  for (int ni = 0; ni < 4; ++ni) {
    const int col = nb + waveN + ni * 16 + rlane;
    const float bv = bias[col];
#pragma unroll
    for (int mi = 0; mi < 2; ++mi) {
#pragma unroll
      for (int i = 0; i < 8; ++i) {
        const int rw = m0 + waveM + mi * 16 + mhi + i;
        C[(size_t)rw * N + col] = acc[mi][ni][i] + bv;
      }
    }
  }
}

// ---------------------------------------------------------------------------
// BatchNorm1d (training-mode batch stats, biased var) + ReLU + bf16 recode.
// grid (N/256, 4): thread owns one column; stats over all rows (redundant per
// y-block, cheap vs GEMM), normalize writes row-chunk [biy*256, biy*256+256).
__global__ __launch_bounds__(256)
void sdn_bn_relu(const float* __restrict__ z, const float* __restrict__ g,
                 const float* __restrict__ be, unsigned short* __restrict__ h,
                 int B, int N) {
  const int c = blockIdx.x * blockDim.x + threadIdx.x;
  if (c >= N) return;
  float s = 0.f, s2 = 0.f;
  for (int r = 0; r < B; ++r) {
    float v = z[(size_t)r * N + c];
    s += v; s2 += v * v;
  }
  const float inv = 1.0f / (float)B;
  const float mean = s * inv;
  const float var  = s2 * inv - mean * mean;
  const float rstd = rsqrtf(var + BN_EPS);
  const float gg = g[c] * rstd;
  const float bb = be[c] - mean * gg;
  const int r0 = blockIdx.y * (B / 4);
  for (int r = r0; r < r0 + B / 4; ++r) {
    float v = z[(size_t)r * N + c] * gg + bb;
    v = v > 0.f ? v : 0.f;
    h[(size_t)r * N + c] = f2bf(v);
  }
}

// ---------------------------------------------------------------------------
extern "C" void kernel_launch(void* const* d_in, const int* in_sizes, int n_in,
                              void* d_out, int out_size, void* d_ws, size_t ws_size,
                              hipStream_t stream) {
  const float* x  = (const float*)d_in[0];
  const float* W1 = (const float*)d_in[1];
  const float* b1 = (const float*)d_in[2];
  const float* W2 = (const float*)d_in[3];
  const float* b2 = (const float*)d_in[4];
  const float* W3 = (const float*)d_in[5];
  const float* b3 = (const float*)d_in[6];
  const float* W4 = (const float*)d_in[7];
  const float* b4 = (const float*)d_in[8];
  const float* W5 = (const float*)d_in[9];
  const float* b5 = (const float*)d_in[10];
  const float* g1 = (const float*)d_in[11]; const float* be1 = (const float*)d_in[12];
  const float* g2 = (const float*)d_in[13]; const float* be2 = (const float*)d_in[14];
  const float* g3 = (const float*)d_in[15]; const float* be3 = (const float*)d_in[16];
  const float* g4 = (const float*)d_in[17]; const float* be4 = (const float*)d_in[18];
  float* out = (float*)d_out;

  char* ws = (char*)d_ws;
  size_t off = 0;
  auto alloc = [&](size_t bytes) {
    void* p = ws + off;
    off += (bytes + 255) & ~(size_t)255;
    return p;
  };
  unsigned short* xp  = (unsigned short*)alloc((size_t)BATCH * DINP * 2);
  unsigned short* W1b = (unsigned short*)alloc((size_t)DINP * DH * 2);
  unsigned short* W2b = (unsigned short*)alloc((size_t)DH * DH * 2);
  unsigned short* W3b = (unsigned short*)alloc((size_t)DH * DH * 2);
  unsigned short* W4b = (unsigned short*)alloc((size_t)DH * DH * 2);
  unsigned short* W5b = (unsigned short*)alloc((size_t)DH * NOUT * 2);
  float*          z   = (float*)         alloc((size_t)BATCH * DH * 4);
  unsigned short* hA  = (unsigned short*)alloc((size_t)BATCH * DH * 2);
  unsigned short* hB  = (unsigned short*)alloc((size_t)BATCH * DH * 2);

  auto nblk = [](long n) { return (unsigned)((n + 255) / 256); };

  // ---- precision conversion: x -> padded bf16 row-major; W -> swizzled bf16 ----
  sdn_cvt_pad_cols<<<nblk((long)BATCH * DINP), 256, 0, stream>>>(x, xp, BATCH, DIN, DINP);
  sdn_cvt_swz<<<dim3(DH / 32, DINP / 32), 256, 0, stream>>>(W1, W1b, DIN, DH);
  sdn_cvt_swz<<<dim3(DH / 32, DH / 32),   256, 0, stream>>>(W2, W2b, DH, DH);
  sdn_cvt_swz<<<dim3(DH / 32, DH / 32),   256, 0, stream>>>(W3, W3b, DH, DH);
  sdn_cvt_swz<<<dim3(DH / 32, DH / 32),   256, 0, stream>>>(W4, W4b, DH, DH);
  sdn_cvt_swz<<<dim3(NOUT / 32, DH / 32), 256, 0, stream>>>(W5, W5b, DH, NOUT);

  const dim3 gemmGridH(DH / 128, BATCH / 128);     // 30 x 8
  const dim3 gemmGridO(NOUT / 128, BATCH / 128);   // 60 x 8
  const dim3 bnGrid(DH / 256, 4);

  // block 1
  sdn_gemm_bf16_wmma<<<gemmGridH, 256, 0, stream>>>(xp, W1b, b1, z, BATCH, DH, DINP);
  sdn_bn_relu<<<bnGrid, 256, 0, stream>>>(z, g1, be1, hA, BATCH, DH);
  // block 2
  sdn_gemm_bf16_wmma<<<gemmGridH, 256, 0, stream>>>(hA, W2b, b2, z, BATCH, DH, DH);
  sdn_bn_relu<<<bnGrid, 256, 0, stream>>>(z, g2, be2, hB, BATCH, DH);
  // block 3
  sdn_gemm_bf16_wmma<<<gemmGridH, 256, 0, stream>>>(hB, W3b, b3, z, BATCH, DH, DH);
  sdn_bn_relu<<<bnGrid, 256, 0, stream>>>(z, g3, be3, hA, BATCH, DH);
  // block 4
  sdn_gemm_bf16_wmma<<<gemmGridH, 256, 0, stream>>>(hA, W4b, b4, z, BATCH, DH, DH);
  sdn_bn_relu<<<bnGrid, 256, 0, stream>>>(z, g4, be4, hB, BATCH, DH);
  // fused 1920 heads: single GEMM -> d_out (fp32)
  sdn_gemm_bf16_wmma<<<gemmGridO, 256, 0, stream>>>(hB, W5b, b5, out, BATCH, NOUT, DH);
}
